// LeaPformerAttention_79164837200633
// MI455X (gfx1250) — matrix operand order repaired and checked by reading, not compiled, not run on hardware
//
#include <hip/hip_runtime.h>
#include <hip/hip_bf16.h>

#define NB 4
#define NH 16
#define NS 4096
#define ND 64
#define NI 32
#define BHN (NB * NH)
#define CHUNK 256
#define CSTR 264   // LDS row stride (elements) for transposed k/V tiles
#define XSTR 72    // LDS row stride (elements) for row-major bf16 tiles
#define HSTR 36    // LDS row stride (floats) for hidden activations
#define HALF_PI_F 1.57079632679489662f

typedef __attribute__((ext_vector_type(16))) __bf16 v16bf;
typedef __attribute__((ext_vector_type(8)))  float v8f;

union FragAB { v16bf v; unsigned int u[8]; };
union FragC  { v8f   v; float        f[8]; };

__device__ __forceinline__ unsigned short f2bf(float f) {
  union { float f; unsigned int u; } x; x.f = f;
  unsigned int u = x.u;
  u += 0x7FFFu + ((u >> 16) & 1u);           // round-to-nearest-even
  return (unsigned short)(u >> 16);
}
__device__ __forceinline__ float bf2f(unsigned short h) {
  union { unsigned int u; float f; } x; x.u = ((unsigned int)h) << 16;
  return x.f;
}

#define WMMA_BF16(Afrag, Bfrag, Cfrag) \
  __builtin_amdgcn_wmma_f32_16x16x32_bf16(false, (Afrag).v, false, (Bfrag).v, \
                                          (short)0, (Cfrag).v, false, false)

__global__ void zero_ws_kernel(float* __restrict__ p, int n) {
  int i = blockIdx.x * blockDim.x + threadIdx.x;
  if (i < n) p[i] = 0.0f;
}

// A-fragment dword offset within a 32-wide K step (16-bit data).
__device__ __forceinline__ int koffA(int j, int lhi) {
  return ((j < 4) ? (2 * j) : (16 + 2 * (j - 4))) + lhi * 8;
}
// B-fragment dword offset within a 32-wide K step (16-bit data).
__device__ __forceinline__ int koffB(int j, int lhi) {
  return ((j < 4) ? (2 * j) : (16 + 2 * (j - 4))) + lhi * 8;
}

// ---------------------------------------------------------------------------
// Kernel 1: per (bh, s-chunk).
//   X = relu(K)*mask  (bf16, LDS row-major)
//   H = X @ W1 + b1   (WMMA)           -> layer2+sigmoid+sin/cos per token
//   inter_{sin,cos} += kfeat^T @ V     (WMMA, atomic reduce to L2)
//   ksum_{sin,cos}  += sum_s kfeat
// ---------------------------------------------------------------------------
__global__ __launch_bounds__(256, 1)
void lea_kv_kernel(const float* __restrict__ K, const float* __restrict__ V,
                   const int* __restrict__ mask,
                   const float* __restrict__ W1, const float* __restrict__ b1,
                   const float* __restrict__ W2, const float* __restrict__ b2,
                   float* __restrict__ inter_sin, float* __restrict__ inter_cos,
                   float* __restrict__ ksum_sin, float* __restrict__ ksum_cos)
{
  __shared__ unsigned short sX  [CHUNK * XSTR];  // relu(K)*mask, [s][d]
  __shared__ float          sH  [CHUNK * HSTR];  // hidden acts,  [s][i]
  __shared__ unsigned short sKs [ND * CSTR];     // k_sin^T [d][s]
  __shared__ unsigned short sKc [ND * CSTR];     // k_cos^T [d][s]
  __shared__ unsigned short sVt [ND * CSTR];     // V^T     [e][s]
  __shared__ unsigned short sW1t[NI * XSTR];     // W1^T bf16 [i][d]
  __shared__ float sB1[NI];
  __shared__ float sW2[NI];
  __shared__ float sB2[1];

  const int t  = threadIdx.x;
  const int bh = blockIdx.x;
  const int b  = bh / NH;
  const int s0 = blockIdx.y * CHUNK;

  // cooperative weight staging: W1 row-major [d][i] -> bf16 W1^T [i][d]
  for (int i = t; i < ND * NI; i += 256) {
    const int d = i >> 5, ii = i & (NI - 1);
    sW1t[ii * XSTR + d] = f2bf(W1[i]);
  }
  if (t < NI) { sB1[t] = b1[t]; sW2[t] = W2[t]; }
  if (t == 0) sB2[0] = b2[0];

  // ---- phase 0: stage relu(K)*mask (row-major) and V^T (col-major) ----
  const size_t rowbase = ((size_t)bh * NS + s0 + t) * ND;
  const float* __restrict__ Krow = K + rowbase;
  const float* __restrict__ Vrow = V + rowbase;
  const float msk = (float)mask[b * NS + s0 + t];

  #pragma unroll 4
  for (int d = 0; d < ND; ++d) {
    sX[t * XSTR + d] = f2bf(fmaxf(Krow[d], 0.0f) * msk);
    sVt[d * CSTR + t] = f2bf(Vrow[d]);
  }
  __syncthreads();

  const int wave = t >> 5, lane = t & 31;
  const int l15 = lane & 15, lhi = lane >> 4;

  // ---- phase 1: H = X @ W1 + b1 via WMMA (16x2 tiles, 4 per wave) ----
  #pragma unroll
  for (int j4 = 0; j4 < 4; ++j4) {
    const int t2 = wave * 4 + j4;
    const int mb = (t2 >> 1) * 16;
    const int nb = (t2 & 1) * 16;
    FragC c;
    const float bias = sB1[nb + l15];
    #pragma unroll
    for (int r = 0; r < 8; ++r) c.f[r] = bias;
    #pragma unroll
    for (int kb = 0; kb < ND; kb += 32) {
      FragAB a, bw;
      const int abase = (mb + l15) * XSTR + kb;
      const int bbase = (nb + l15) * XSTR + kb;
      #pragma unroll
      for (int j = 0; j < 8; ++j) {
        a.u[j]  = *(const unsigned int*)&sX[abase + koffA(j, lhi)];
        bw.u[j] = *(const unsigned int*)&sW1t[bbase + koffB(j, lhi)];
      }
      c.v = WMMA_BF16(a, bw, c);
    }
    #pragma unroll
    for (int r = 0; r < 8; ++r)
      sH[(mb + r + lhi * 8) * HSTR + nb + l15] = c.f[r];
  }
  __syncthreads();

  // ---- phase 2: per-token layer2 + sigmoid + sin/cos, emit features ----
  {
    float z = sB2[0];
    const float* __restrict__ hrow = &sH[t * HSTR];
    #pragma unroll
    for (int i = 0; i < NI; ++i) z = fmaf(fmaxf(hrow[i], 0.0f), sW2[i], z);
    const float num = 1.0f / (1.0f + __expf(-z));
    const float ang = HALF_PI_F * num;
    const float sv = __sinf(ang), cv = __cosf(ang);
    #pragma unroll 4
    for (int d = 0; d < ND; ++d) {
      const float xd = bf2f(sX[t * XSTR + d]);
      sKs[d * CSTR + t] = f2bf(sv * xd);
      sKc[d * CSTR + t] = f2bf(cv * xd);
    }
  }
  __syncthreads();

  // ---- phase 3: inter += kfeat^T @ V (16 tiles of 64x64; 2 per wave) ----
  FragC cs[2], cc[2];
  #pragma unroll
  for (int ti = 0; ti < 2; ++ti)
    #pragma unroll
    for (int r = 0; r < 8; ++r) { cs[ti].f[r] = 0.0f; cc[ti].f[r] = 0.0f; }

  #pragma unroll 1
  for (int kb = 0; kb < CHUNK; kb += 32) {
    #pragma unroll
    for (int ti = 0; ti < 2; ++ti) {
      const int tile = wave * 2 + ti;
      const int mb = (tile >> 2) * 16;
      const int nb = (tile & 3) * 16;
      FragAB as, ac, bb;
      const int abase = (mb + l15) * CSTR + kb;
      const int bbase = (nb + l15) * CSTR + kb;
      #pragma unroll
      for (int j = 0; j < 8; ++j) {
        as.u[j] = *(const unsigned int*)&sKs[abase + koffA(j, lhi)];
        ac.u[j] = *(const unsigned int*)&sKc[abase + koffA(j, lhi)];
        bb.u[j] = *(const unsigned int*)&sVt[bbase + koffB(j, lhi)];
      }
      cs[ti].v = WMMA_BF16(as, bb, cs[ti]);
      cc[ti].v = WMMA_BF16(ac, bb, cc[ti]);
    }
  }

  float* __restrict__ gs = inter_sin + (size_t)bh * (ND * ND);
  float* __restrict__ gc = inter_cos + (size_t)bh * (ND * ND);
  #pragma unroll
  for (int ti = 0; ti < 2; ++ti) {
    const int tile = wave * 2 + ti;
    const int mb = (tile >> 2) * 16;
    const int nb = (tile & 3) * 16;
    #pragma unroll
    for (int r = 0; r < 8; ++r) {
      const int m = mb + r + lhi * 8;
      const int n = nb + l15;
      atomicAdd(&gs[m * ND + n], cs[ti].f[r]);
      atomicAdd(&gc[m * ND + n], cc[ti].f[r]);
    }
  }

  // ---- ksum: threads 0..63 sum k_sin rows, 64..127 sum k_cos rows ----
  if (t < 2 * ND) {
    const int d = t & (ND - 1);
    const unsigned short* __restrict__ row = (t < ND) ? &sKs[d * CSTR] : &sKc[d * CSTR];
    float acc = 0.0f;
    #pragma unroll 4
    for (int s = 0; s < CHUNK; ++s) acc += bf2f(row[s]);
    float* dst = (t < ND) ? &ksum_sin[bh * ND + d] : &ksum_cos[bh * ND + d];
    atomicAdd(dst, acc);
  }
}

// ---------------------------------------------------------------------------
// Kernel 2: per (bh, s-chunk).
//   X = relu(Q); H = X @ W1 + b1 (WMMA); sigmoid+sin/cos per token;
//   out = (qfeat @ inter) / (qfeat . ksum)   (WMMA)
// ---------------------------------------------------------------------------
__global__ __launch_bounds__(256, 1)
void lea_q_kernel(const float* __restrict__ Q,
                  const float* __restrict__ W1, const float* __restrict__ b1,
                  const float* __restrict__ W2, const float* __restrict__ b2,
                  const float* __restrict__ inter_sin, const float* __restrict__ inter_cos,
                  const float* __restrict__ ksum_sin, const float* __restrict__ ksum_cos,
                  float* __restrict__ out)
{
  __shared__ unsigned short sX  [CHUNK * XSTR];  // relu(Q) [s][d]
  __shared__ float          sH  [CHUNK * HSTR];  // hidden acts [s][i]
  __shared__ unsigned short sQs [CHUNK * XSTR];  // q_sin [s][d] (A frag order)
  __shared__ unsigned short sQc [CHUNK * XSTR];  // q_cos [s][d]
  __shared__ unsigned short sIs [ND * XSTR];     // inter_sin^T [e][d] (B frag order)
  __shared__ unsigned short sIc [ND * XSTR];     // inter_cos^T [e][d]
  __shared__ unsigned short sW1t[NI * XSTR];     // W1^T bf16 [i][d]
  __shared__ float sNorm[CHUNK];
  __shared__ float sSs[ND];
  __shared__ float sSc[ND];
  __shared__ float sB1[NI];
  __shared__ float sW2[NI];
  __shared__ float sB2[1];

  const int t  = threadIdx.x;
  const int bh = blockIdx.x;
  const int s0 = blockIdx.y * CHUNK;

  for (int i = t; i < ND * NI; i += 256) {
    const int d = i >> 5, ii = i & (NI - 1);
    sW1t[ii * XSTR + d] = f2bf(W1[i]);
  }
  if (t < NI) { sB1[t] = b1[t]; sW2[t] = W2[t]; }
  if (t == 0) sB2[0] = b2[0];
  if (t < ND) { sSs[t] = ksum_sin[bh * ND + t]; sSc[t] = ksum_cos[bh * ND + t]; }
  for (int i = t; i < ND * ND; i += 256) {
    const int d = i >> 6, e = i & 63;
    sIs[e * XSTR + d] = f2bf(inter_sin[(size_t)bh * ND * ND + i]);
    sIc[e * XSTR + d] = f2bf(inter_cos[(size_t)bh * ND * ND + i]);
  }

  // ---- phase 0: stage relu(Q) row-major ----
  const float* __restrict__ Qrow = Q + ((size_t)bh * NS + s0 + t) * ND;
  #pragma unroll 4
  for (int d = 0; d < ND; ++d)
    sX[t * XSTR + d] = f2bf(fmaxf(Qrow[d], 0.0f));
  __syncthreads();

  const int wave = t >> 5, lane = t & 31;
  const int l15 = lane & 15, lhi = lane >> 4;

  // ---- phase 1: H = X @ W1 + b1 via WMMA ----
  #pragma unroll
  for (int j4 = 0; j4 < 4; ++j4) {
    const int t2 = wave * 4 + j4;
    const int mb = (t2 >> 1) * 16;
    const int nb = (t2 & 1) * 16;
    FragC c;
    const float bias = sB1[nb + l15];
    #pragma unroll
    for (int r = 0; r < 8; ++r) c.f[r] = bias;
    #pragma unroll
    for (int kb = 0; kb < ND; kb += 32) {
      FragAB a, bw;
      const int abase = (mb + l15) * XSTR + kb;
      const int bbase = (nb + l15) * XSTR + kb;
      #pragma unroll
      for (int j = 0; j < 8; ++j) {
        a.u[j]  = *(const unsigned int*)&sX[abase + koffA(j, lhi)];
        bw.u[j] = *(const unsigned int*)&sW1t[bbase + koffB(j, lhi)];
      }
      c.v = WMMA_BF16(a, bw, c);
    }
    #pragma unroll
    for (int r = 0; r < 8; ++r)
      sH[(mb + r + lhi * 8) * HSTR + nb + l15] = c.f[r];
  }
  __syncthreads();

  // ---- phase 2: layer2 + sigmoid + sin/cos, features + norm ----
  {
    float z = sB2[0];
    const float* __restrict__ hrow = &sH[t * HSTR];
    #pragma unroll
    for (int i = 0; i < NI; ++i) z = fmaf(fmaxf(hrow[i], 0.0f), sW2[i], z);
    const float num = 1.0f / (1.0f + __expf(-z));
    const float ang = HALF_PI_F * num;
    const float sv = __sinf(ang), cv = __cosf(ang);
    float nacc = 0.0f;
    #pragma unroll 4
    for (int d = 0; d < ND; ++d) {
      const float xd = bf2f(sX[t * XSTR + d]);
      const float qs = sv * xd, qc = cv * xd;
      sQs[t * XSTR + d] = f2bf(qs);
      sQc[t * XSTR + d] = f2bf(qc);
      nacc = fmaf(qs, sSs[d], fmaf(qc, sSc[d], nacc));
    }
    sNorm[t] = nacc;
  }
  __syncthreads();

  // ---- phase 3: out tile = (qfeat @ inter) / norm; 64 tiles, 8 per wave ----
  float* __restrict__ outBase = out + ((size_t)bh * NS + s0) * ND;
  #pragma unroll 1
  for (int ti = 0; ti < 8; ++ti) {
    const int tile = wave * 8 + ti;
    const int mb = (tile >> 2) * 16;
    const int nb = (tile & 3) * 16;
    FragC c;
    #pragma unroll
    for (int r = 0; r < 8; ++r) c.f[r] = 0.0f;

    #pragma unroll
    for (int kb = 0; kb < ND; kb += 32) {
      FragAB aS, aC, bS, bC;
      const int abase = (mb + l15) * XSTR + kb;
      const int bbase = (nb + l15) * XSTR + kb;
      #pragma unroll
      for (int j = 0; j < 8; ++j) {
        aS.u[j] = *(const unsigned int*)&sQs[abase + koffA(j, lhi)];
        aC.u[j] = *(const unsigned int*)&sQc[abase + koffA(j, lhi)];
        bS.u[j] = *(const unsigned int*)&sIs[bbase + koffB(j, lhi)];
        bC.u[j] = *(const unsigned int*)&sIc[bbase + koffB(j, lhi)];
      }
      c.v = WMMA_BF16(aS, bS, c);
      c.v = WMMA_BF16(aC, bC, c);
    }

    #pragma unroll
    for (int r = 0; r < 8; ++r) {
      const int row = mb + r + lhi * 8;
      const float inv = 1.0f / sNorm[row];
      outBase[(size_t)row * ND + nb + l15] = c.f[r] * inv;
    }
  }
}

// ---------------------------------------------------------------------------
extern "C" void kernel_launch(void* const* d_in, const int* in_sizes, int n_in,
                              void* d_out, int out_size, void* d_ws, size_t ws_size,
                              hipStream_t stream) {
  (void)in_sizes; (void)n_in; (void)out_size; (void)ws_size;
  const float* Q    = (const float*)d_in[0];
  const float* K    = (const float*)d_in[1];
  const float* V    = (const float*)d_in[2];
  const int*   mask = (const int*)d_in[3];
  const float* qW1  = (const float*)d_in[4];
  const float* qb1  = (const float*)d_in[5];
  const float* qW2  = (const float*)d_in[6];
  const float* qb2  = (const float*)d_in[7];
  const float* kW1  = (const float*)d_in[8];
  const float* kb1  = (const float*)d_in[9];
  const float* kW2  = (const float*)d_in[10];
  const float* kb2  = (const float*)d_in[11];

  float* ws        = (float*)d_ws;
  float* inter_sin = ws;
  float* inter_cos = inter_sin + (size_t)BHN * ND * ND;
  float* ksum_sin  = inter_cos + (size_t)BHN * ND * ND;
  float* ksum_cos  = ksum_sin + (size_t)BHN * ND;

  const int nzero = BHN * ND * ND * 2 + BHN * ND * 2;
  zero_ws_kernel<<<(nzero + 255) / 256, 256, 0, stream>>>(ws, nzero);

  dim3 grid(BHN, NS / CHUNK);
  lea_kv_kernel<<<grid, 256, 0, stream>>>(K, V, mask, kW1, kb1, kW2, kb2,
                                          inter_sin, inter_cos, ksum_sin, ksum_cos);
  lea_q_kernel<<<grid, 256, 0, stream>>>(Q, qW1, qb1, qW2, qb2,
                                         inter_sin, inter_cos, ksum_sin, ksum_cos,
                                         (float*)d_out);
}
